// Path_Complex_layer_11484742549814
// MI455X (gfx1250) — matrix operand order repaired and checked by reading, not compiled, not run on hardware
//
#include <hip/hip_runtime.h>
#include <hip/hip_bf16.h>

// ---------------------------------------------------------------------------
// MI455X / gfx1250 implementation.
//  - All dense GEMMs (X [M,K] * W[Nout,K]^T) run on v_wmma_f32_16x16x32_bf16.
//    Each wave32 computes a 16(M) x 128(N) strip with 8 f32 accumulators:
//    the A fragment is register-reused across 8 WMMAs per K-step, cutting
//    redundant A HBM traffic from x16 to x2 vs naive 1-tile-per-wave.
//    K is a template parameter so all B-tile/K-step addresses become
//    immediate offsets and the K loop fully unrolls.
//  - Graph scatter / edge-softmax stages are coalesced fp32 kernels with
//    L2-resident atomics (accumulator tables ~164MB vs 192MB L2).
// ---------------------------------------------------------------------------

typedef __attribute__((ext_vector_type(16))) __bf16 v16bf;
typedef __attribute__((ext_vector_type(8)))  float  v8f;

#define LRELU(x) ((x) > 0.0f ? (x) : 0.01f * (x))

__device__ __forceinline__ unsigned short f2bf(float f) {
  unsigned int u = __float_as_uint(f);
  u += 0x7FFFu + ((u >> 16) & 1u);       // round-to-nearest-even
  return (unsigned short)(u >> 16);
}

// float atomic-max via signed-max / unsigned-min monotonic bit trick
__device__ __forceinline__ void atomicMaxF(float* a, float v) {
  if (v >= 0.0f) atomicMax((int*)a, __float_as_int(v));
  else           atomicMin((unsigned int*)a, (unsigned int)__float_as_int(v));
}

// ---------------------------- utility kernels ------------------------------
__global__ void k_f32_to_bf16(const float* __restrict__ in,
                              unsigned short* __restrict__ out, long long n) {
  long long i = (long long)blockIdx.x * blockDim.x + threadIdx.x;
  if (i < n) out[i] = f2bf(in[i]);
}

__global__ void k_fill(float* __restrict__ p, float v, long long n) {
  long long i = (long long)blockIdx.x * blockDim.x + threadIdx.x;
  if (i < n) p[i] = v;
}

// ------------------------------- WMMA GEMM ---------------------------------
// C[M,Nout] (f32, row-major) = A[M,K] (bf16 rm) * B[Nout,K]^T (bf16 rm) + bias
// Wave strip: 16 rows x 128 cols (8 accumulators). Block = 4 waves = 64 rows.
// grid = (ceil(M/64), Nout/128).
// A fragment: lane half h, row r=lane&15:
//   elements 0..15 = K = k0 + 16h .. +15    (32B contiguous)
// B fragment: row n, elements 0..15 = K = k0 + 16h .. +15 (32B contiguous)
template <int K>
__global__ void gemm_bf16_nt(const unsigned short* __restrict__ A,
                             const unsigned short* __restrict__ B,
                             const float* __restrict__ bias,
                             float* __restrict__ C,
                             int M, int Nout) {
  const int wave = threadIdx.x >> 5;
  const int lane = threadIdx.x & 31;
  const int m0 = blockIdx.x * 64 + wave * 16;
  const int n0 = blockIdx.y * 128;
  if (m0 >= M) return;                        // wave-uniform: EXEC stays full

  const int half = lane >> 4;
  const int r    = lane & 15;

  const uint4* __restrict__ arow =
      (const uint4*)(A + (size_t)(m0 + r) * K);
  const unsigned short* __restrict__ bbase = B + (size_t)(n0 + r) * K;

  v8f acc[8] = {};
  union Frag { v16bf v; uint4 u[2]; };

#pragma unroll
  for (int k0 = 0; k0 < K; k0 += 32) {
    Frag fa;
    const int ca = (k0 >> 3) + half * 2;      // A: 16 contiguous K per half
    fa.u[0] = arow[ca];
    fa.u[1] = arow[ca + 1];
    const int cb = (k0 >> 3) + half * 2;      // B: 16 contiguous K per half
#pragma unroll
    for (int nt = 0; nt < 8; ++nt) {
      Frag fb;
      const uint4* bp = (const uint4*)(bbase + (size_t)nt * 16 * K);
      fb.u[0] = bp[cb];
      fb.u[1] = bp[cb + 1];
      acc[nt] = __builtin_amdgcn_wmma_f32_16x16x32_bf16(
          false, fa.v, false, fb.v, (short)0, acc[nt], false, false);
    }
  }

#pragma unroll
  for (int nt = 0; nt < 8; ++nt) {
    const int col = n0 + nt * 16 + r;
    const float bv = bias ? bias[col] : 0.0f;
#pragma unroll
    for (int j = 0; j < 8; ++j) {             // VGPR j -> row j + 8*half
      C[(size_t)(m0 + j + half * 8) * Nout + col] = acc[nt][j] + bv;
    }
  }
}

// --------------------- attention logits (one wave / edge) ------------------
// elog[e,h] = sum_d lrelu(ni[src[e],h*64+d] + nj[dst[e],..] + fij[e,..] +
//                         bias[..]) * attn[h*64+d]
__global__ void k_attn_logits(const float* __restrict__ ni,
                              const float* __restrict__ nj,
                              const float* __restrict__ fij,
                              const int* __restrict__ src,
                              const int* __restrict__ dst,
                              const float* __restrict__ bias,
                              const float* __restrict__ attn,
                              float* __restrict__ elog, int nE) {
  const int e = blockIdx.x * (blockDim.x >> 5) + (threadIdx.x >> 5);
  if (e >= nE) return;
  const int lane = threadIdx.x & 31;

  const float4* pni = (const float4*)(ni  + (size_t)src[e] * 256);
  const float4* pnj = (const float4*)(nj  + (size_t)dst[e] * 256);
  const float4* pf  = (const float4*)(fij + (size_t)e      * 256);
  const float4* pb  = (const float4*)bias;
  const float4* pa  = (const float4*)attn;

  float s = 0.0f;
#pragma unroll
  for (int q = 0; q < 2; ++q) {
    const int c4 = lane * 2 + q;              // channels lane*8 .. lane*8+7
    float4 a = pni[c4], b = pnj[c4], f = pf[c4], bb = pb[c4], at = pa[c4];
    float t;
    t = a.x + b.x + f.x + bb.x; t = LRELU(t); s += t * at.x;
    t = a.y + b.y + f.y + bb.y; t = LRELU(t); s += t * at.y;
    t = a.z + b.z + f.z + bb.z; t = LRELU(t); s += t * at.z;
    t = a.w + b.w + f.w + bb.w; t = LRELU(t); s += t * at.w;
  }
  // reduce within each 8-lane group (one head per group)
  s += __shfl_xor(s, 1, 32);
  s += __shfl_xor(s, 2, 32);
  s += __shfl_xor(s, 4, 32);
  if ((lane & 7) == 0) elog[(size_t)e * 4 + (lane >> 3)] = s;
}

// ------------------------- edge softmax (segment) --------------------------
__global__ void k_seg_max(const float* __restrict__ elog,
                          const int* __restrict__ dst,
                          float* __restrict__ mx, int nE) {
  long long i = (long long)blockIdx.x * blockDim.x + threadIdx.x;
  if (i >= (long long)nE * 4) return;
  const int e = (int)(i >> 2), h = (int)(i & 3);
  atomicMaxF(&mx[(size_t)dst[e] * 4 + h], elog[i]);
}

__global__ void k_seg_expsum(float* __restrict__ elog,
                             const int* __restrict__ dst,
                             const float* __restrict__ mx,
                             float* __restrict__ den, int nE) {
  long long i = (long long)blockIdx.x * blockDim.x + threadIdx.x;
  if (i >= (long long)nE * 4) return;
  const int e = (int)(i >> 2), h = (int)(i & 3);
  const float ex = __expf(elog[i] - mx[(size_t)dst[e] * 4 + h]);
  elog[i] = ex;
  atomicAdd(&den[(size_t)dst[e] * 4 + h], ex);
}

__global__ void k_seg_norm(float* __restrict__ elog,
                           const int* __restrict__ dst,
                           const float* __restrict__ den, int nE) {
  long long i = (long long)blockIdx.x * blockDim.x + threadIdx.x;
  if (i >= (long long)nE * 4) return;
  const int e = (int)(i >> 2), h = (int)(i & 3);
  elog[i] = elog[i] / den[(size_t)dst[e] * 4 + h];
}

// ------------------- mean aggregation of x_feats onto dst ------------------
__global__ void k_count_agg(const float* __restrict__ x,
                            const int* __restrict__ dst,
                            float* __restrict__ cnt,
                            float* __restrict__ agg, int nE2) {
  long long i = (long long)blockIdx.x * blockDim.x + threadIdx.x;
  if (i >= (long long)nE2 * 128) return;
  const int e = (int)(i >> 7), c = (int)(i & 127);
  const int d = dst[e];
  atomicAdd(&agg[(size_t)d * 128 + c], x[i]);
  if (c == 0) atomicAdd(&cnt[d], 1.0f);
}

// m1 = concat(m_feats, agg/max(cnt,1)) -> bf16
__global__ void k_build_m1(const float* __restrict__ m,
                           const float* __restrict__ agg,
                           const float* __restrict__ cnt,
                           unsigned short* __restrict__ m1, int nE) {
  long long i = (long long)blockIdx.x * blockDim.x + threadIdx.x;
  if (i >= (long long)nE * 256) return;
  const int e = (int)(i >> 8), c = (int)(i & 255);
  float v;
  if (c < 128) v = m[(size_t)e * 128 + c];
  else         v = agg[(size_t)e * 128 + (c - 128)] / fmaxf(cnt[e], 1.0f);
  m1[i] = f2bf(v);
}

// m2 = concat(l[src]+l[dst], m_feats) -> bf16
__global__ void k_build_m2(const float* __restrict__ l,
                           const float* __restrict__ m,
                           const int* __restrict__ src,
                           const int* __restrict__ dst,
                           unsigned short* __restrict__ m2, int nE) {
  long long i = (long long)blockIdx.x * blockDim.x + threadIdx.x;
  if (i >= (long long)nE * 256) return;
  const int e = (int)(i >> 8), c = (int)(i & 255);
  float v;
  if (c < 128) v = l[(size_t)src[e] * 128 + c] + l[(size_t)dst[e] * 128 + c];
  else         v = m[(size_t)e * 128 + (c - 128)];
  m2[i] = f2bf(v);
}

// hdst[dst[e2], c] += h[src[e2], c] * a[(aFromDst ? dst[e2] : e2), c/64]
__global__ void k_scatter_edge(const float* __restrict__ h,
                               const float* __restrict__ a,
                               const int* __restrict__ src,
                               const int* __restrict__ dst,
                               float* __restrict__ hdst,
                               int nE2, int aFromDst) {
  long long i = (long long)blockIdx.x * blockDim.x + threadIdx.x;
  if (i >= (long long)nE2 * 256) return;
  const int e2 = (int)(i >> 8), c = (int)(i & 255), hh = c >> 6;
  const int s = src[e2], d = dst[e2];
  const float av = a[(size_t)(aFromDst ? d : e2) * 4 + hh];
  atomicAdd(&hdst[(size_t)d * 256 + c], h[(size_t)s * 256 + c] * av);
}

// out[e,d] = (addin ? addin[e,d] : 0) + sum_h lrelu(hdst[e, h*64+d])
__global__ void k_head_reduce(const float* __restrict__ hdst,
                              const float* __restrict__ addin,
                              float* __restrict__ out, int nE) {
  long long i = (long long)blockIdx.x * blockDim.x + threadIdx.x;
  if (i >= (long long)nE * 64) return;
  const int e = (int)(i >> 6), d = (int)(i & 63);
  const float* p = hdst + (size_t)e * 256 + d;
  float s = LRELU(p[0]) + LRELU(p[64]) + LRELU(p[128]) + LRELU(p[192]);
  if (addin) s += addin[i];
  out[i] = s;
}

// ---------------------------------------------------------------------------
static inline dim3 grid1(long long n) {
  return dim3((unsigned)((n + 255) / 256));
}

extern "C" void kernel_launch(void* const* d_in, const int* in_sizes, int n_in,
                              void* d_out, int out_size, void* d_ws, size_t ws_size,
                              hipStream_t stream) {
  const float* l_feats   = (const float*)d_in[0];
  const float* m_feats   = (const float*)d_in[1];
  const float* x_feats   = (const float*)d_in[2];
  const int*   g_src     = (const int*)d_in[3];
  const int*   g_dst     = (const int*)d_in[4];
  const int*   lg_src    = (const int*)d_in[5];
  const int*   lg_dst    = (const int*)d_in[6];
  const float* W_lg_node = (const float*)d_in[7];
  const float* b_lg_node = (const float*)d_in[8];
  const float* W_lg_ni   = (const float*)d_in[9];
  const float* W_lg_fij  = (const float*)d_in[10];
  const float* W_lg_nj   = (const float*)d_in[11];
  const float* lg_attn   = (const float*)d_in[12];
  const float* bias_lg   = (const float*)d_in[13];
  const float* W_g_node  = (const float*)d_in[14];
  const float* b_g_node  = (const float*)d_in[15];
  const float* W_g_ni    = (const float*)d_in[16];
  const float* W_g_fij   = (const float*)d_in[17];
  const float* W_g_nj    = (const float*)d_in[18];
  const float* g_attn    = (const float*)d_in[19];
  const float* bias_g    = (const float*)d_in[20];

  const int Nn = in_sizes[0] / 128;   // graph nodes
  const int E  = in_sizes[1] / 128;   // graph edges == line-graph nodes
  const int E2 = in_sizes[2] / 128;   // line-graph edges

  // ---------------- workspace carve (256B aligned) ----------------
  char* ws = (char*)d_ws;
  size_t off = 0;
  auto carve = [&](size_t bytes) -> void* {
    off = (off + 255) & ~(size_t)255;
    void* p = ws + off;
    off += bytes;
    return p;
  };

  unsigned short* m_bf   = (unsigned short*)carve((size_t)E  * 128 * 2);
  unsigned short* x_bf   = (unsigned short*)carve((size_t)E2 * 128 * 2);
  unsigned short* l_bf   = (unsigned short*)carve((size_t)Nn * 128 * 2);
  unsigned short* wlni   = (unsigned short*)carve(256 * 128 * 2);
  unsigned short* wlnj   = (unsigned short*)carve(256 * 128 * 2);
  unsigned short* wlfij  = (unsigned short*)carve(256 * 128 * 2);
  unsigned short* wgni   = (unsigned short*)carve(256 * 128 * 2);
  unsigned short* wgnj   = (unsigned short*)carve(256 * 128 * 2);
  unsigned short* wgfij  = (unsigned short*)carve(256 * 128 * 2);
  unsigned short* wlnode = (unsigned short*)carve(256 * 256 * 2);
  unsigned short* wgnode = (unsigned short*)carve(256 * 256 * 2);
  unsigned short* m12_bf = (unsigned short*)carve((size_t)E * 256 * 2); // m1, later m2

  float* big_ni  = (float*)carve((size_t)E  * 256 * 4);  // lg_ni, later g_fij
  float* big_nj  = (float*)carve((size_t)E  * 256 * 4);  // lg_nj
  float* big_fij = (float*)carve((size_t)E2 * 256 * 4);  // lg_fij
  float* h1      = (float*)carve((size_t)E  * 256 * 4);  // h1, later h2
  float* hdst    = (float*)carve((size_t)E  * 256 * 4);  // h1_dst, later h2_dst
  float* lg_out  = (float*)carve((size_t)E  * 64  * 4);
  float* elog2   = (float*)carve((size_t)E2 * 4   * 4);  // lg logits -> a_lg
  float* elogE   = (float*)carve((size_t)E  * 4   * 4);  // g logits  -> a_g
  float* mxE     = (float*)carve((size_t)E  * 4   * 4);
  float* denE    = (float*)carve((size_t)E  * 4   * 4);
  float* cnt     = (float*)carve((size_t)E  * 4);
  float* aggb    = (float*)carve((size_t)E  * 128 * 4);
  float* mxN     = (float*)carve((size_t)Nn * 4   * 4);
  float* denN    = (float*)carve((size_t)Nn * 4   * 4);
  float* gni     = (float*)carve((size_t)Nn * 256 * 4);
  float* gnj     = (float*)carve((size_t)Nn * 256 * 4);
  (void)ws_size; (void)n_in; (void)out_size;

  const dim3 TPB(256);
  const dim3 GB(128);   // 4 waves: 64(M) x 128(N) per block

  // ---------------- precision conversion ----------------
  k_f32_to_bf16<<<grid1((long long)E  * 128), TPB, 0, stream>>>(m_feats, m_bf, (long long)E  * 128);
  k_f32_to_bf16<<<grid1((long long)E2 * 128), TPB, 0, stream>>>(x_feats, x_bf, (long long)E2 * 128);
  k_f32_to_bf16<<<grid1((long long)Nn * 128), TPB, 0, stream>>>(l_feats, l_bf, (long long)Nn * 128);
  k_f32_to_bf16<<<grid1(256 * 128), TPB, 0, stream>>>(W_lg_ni,  wlni,  256 * 128);
  k_f32_to_bf16<<<grid1(256 * 128), TPB, 0, stream>>>(W_lg_nj,  wlnj,  256 * 128);
  k_f32_to_bf16<<<grid1(256 * 128), TPB, 0, stream>>>(W_lg_fij, wlfij, 256 * 128);
  k_f32_to_bf16<<<grid1(256 * 128), TPB, 0, stream>>>(W_g_ni,   wgni,  256 * 128);
  k_f32_to_bf16<<<grid1(256 * 128), TPB, 0, stream>>>(W_g_nj,   wgnj,  256 * 128);
  k_f32_to_bf16<<<grid1(256 * 128), TPB, 0, stream>>>(W_g_fij,  wgfij, 256 * 128);
  k_f32_to_bf16<<<grid1(256 * 256), TPB, 0, stream>>>(W_lg_node, wlnode, 256 * 256);
  k_f32_to_bf16<<<grid1(256 * 256), TPB, 0, stream>>>(W_g_node,  wgnode, 256 * 256);

  // ---------------- line-graph branch ----------------
  gemm_bf16_nt<128><<<dim3((E  + 63) / 64, 2), GB, 0, stream>>>(m_bf, wlni,  nullptr, big_ni,  E,  256);
  gemm_bf16_nt<128><<<dim3((E  + 63) / 64, 2), GB, 0, stream>>>(m_bf, wlnj,  nullptr, big_nj,  E,  256);
  gemm_bf16_nt<128><<<dim3((E2 + 63) / 64, 2), GB, 0, stream>>>(x_bf, wlfij, nullptr, big_fij, E2, 256);

  k_attn_logits<<<dim3((E2 + 7) / 8), TPB, 0, stream>>>(
      big_ni, big_nj, big_fij, lg_src, lg_dst, bias_lg, lg_attn, elog2, E2);

  k_fill<<<grid1((long long)E * 4), TPB, 0, stream>>>(mxE, -1e30f, (long long)E * 4);
  k_fill<<<grid1((long long)E * 4), TPB, 0, stream>>>(denE, 0.0f,  (long long)E * 4);
  k_seg_max   <<<grid1((long long)E2 * 4), TPB, 0, stream>>>(elog2, lg_dst, mxE, E2);
  k_seg_expsum<<<grid1((long long)E2 * 4), TPB, 0, stream>>>(elog2, lg_dst, mxE, denE, E2);
  k_seg_norm  <<<grid1((long long)E2 * 4), TPB, 0, stream>>>(elog2, lg_dst, denE, E2);  // a_lg

  k_fill<<<grid1((long long)E), TPB, 0, stream>>>(cnt, 0.0f, (long long)E);
  k_fill<<<grid1((long long)E * 128), TPB, 0, stream>>>(aggb, 0.0f, (long long)E * 128);
  k_count_agg<<<grid1((long long)E2 * 128), TPB, 0, stream>>>(x_feats, lg_dst, cnt, aggb, E2);
  k_build_m1<<<grid1((long long)E * 256), TPB, 0, stream>>>(m_feats, aggb, cnt, m12_bf, E);

  gemm_bf16_nt<256><<<dim3((E + 63) / 64, 2), GB, 0, stream>>>(m12_bf, wlnode, b_lg_node, h1, E, 256);

  k_fill<<<grid1((long long)E * 256), TPB, 0, stream>>>(hdst, 0.0f, (long long)E * 256);
  k_scatter_edge<<<grid1((long long)E2 * 256), TPB, 0, stream>>>(h1, elog2, lg_src, lg_dst, hdst, E2, 0);
  k_head_reduce<<<grid1((long long)E * 64), TPB, 0, stream>>>(hdst, nullptr, lg_out, E);

  // ---------------- base-graph branch ----------------
  gemm_bf16_nt<128><<<dim3((Nn + 63) / 64, 2), GB, 0, stream>>>(l_bf, wgni, nullptr, gni, Nn, 256);
  gemm_bf16_nt<128><<<dim3((Nn + 63) / 64, 2), GB, 0, stream>>>(l_bf, wgnj, nullptr, gnj, Nn, 256);
  gemm_bf16_nt<128><<<dim3((E  + 63) / 64, 2), GB, 0, stream>>>(m_bf, wgfij, nullptr, big_ni, E, 256); // g_fij

  k_attn_logits<<<dim3((E + 7) / 8), TPB, 0, stream>>>(
      gni, gnj, big_ni, g_src, g_dst, bias_g, g_attn, elogE, E);

  k_fill<<<grid1((long long)Nn * 4), TPB, 0, stream>>>(mxN, -1e30f, (long long)Nn * 4);
  k_fill<<<grid1((long long)Nn * 4), TPB, 0, stream>>>(denN, 0.0f,  (long long)Nn * 4);
  k_seg_max   <<<grid1((long long)E * 4), TPB, 0, stream>>>(elogE, g_dst, mxN, E);
  k_seg_expsum<<<grid1((long long)E * 4), TPB, 0, stream>>>(elogE, g_dst, mxN, denN, E);
  k_seg_norm  <<<grid1((long long)E * 4), TPB, 0, stream>>>(elogE, g_dst, denN, E);    // a_g

  k_build_m2<<<grid1((long long)E * 256), TPB, 0, stream>>>(l_feats, m_feats, g_src, g_dst, m12_bf, E);
  gemm_bf16_nt<256><<<dim3((E + 63) / 64, 2), GB, 0, stream>>>(m12_bf, wgnode, b_g_node, h1, E, 256); // h2

  k_fill<<<grid1((long long)E * 256), TPB, 0, stream>>>(hdst, 0.0f, (long long)E * 256);
  k_scatter_edge<<<grid1((long long)E2 * 256), TPB, 0, stream>>>(h1, elogE, lg_src, lg_dst, hdst, E2, 1);
  k_head_reduce<<<grid1((long long)E * 64), TPB, 0, stream>>>(hdst, lg_out, (float*)d_out, E);
}
